// AttentionDecoder_26834955665971
// MI455X (gfx1250) — matrix operand order repaired and checked by reading, not compiled, not run on hardware
//
#include <hip/hip_runtime.h>
#include <hip/hip_bf16.h>
#include <math.h>

#define DIM   1024
#define BB    64
#define SS    512
#define VOCAB 32000

typedef __attribute__((ext_vector_type(16))) __bf16 v16bf;
typedef __attribute__((ext_vector_type(8)))  float  v8f;

union BF16x16 {
  v16bf v;
  unsigned short u[16];
  uint4 q[2];
};

__device__ __forceinline__ unsigned short f2bf(float f) {
  unsigned int u = __float_as_uint(f);
  unsigned int r = u + 0x7FFFu + ((u >> 16) & 1u);   // round-to-nearest-even
  return (unsigned short)(r >> 16);
}

// gfx1250 transcendental tanh; pad with V_NOPs per CDNA5 trans-op co-exec hazard rule
__device__ __forceinline__ float fast_tanh(float x) {
  float r;
  asm volatile("v_tanh_f32 %0, %1\n\tv_nop\n\tv_nop" : "=v"(r) : "v"(x));
  return r;
}

__device__ __forceinline__ float sigmoidf(float x) {
  return 1.0f / (1.0f + __expf(-x));
}

__device__ __forceinline__ float wave_reduce_add(float v) {
  v += __shfl_xor(v, 16, 32);
  v += __shfl_xor(v, 8, 32);
  v += __shfl_xor(v, 4, 32);
  v += __shfl_xor(v, 2, 32);
  v += __shfl_xor(v, 1, 32);
  return v;
}

// ---------------- W1 f32 -> bf16 (row-major [e][k]) ----------------
__global__ void k_cvt_w1(const float* __restrict__ W, unsigned short* __restrict__ out) {
  int i = (blockIdx.x * 256 + threadIdx.x) * 4;
  float4 f = *(const float4*)(W + i);
  uint2 p;
  p.x = (unsigned)f2bf(f.x) | ((unsigned)f2bf(f.y) << 16);
  p.y = (unsigned)f2bf(f.z) | ((unsigned)f2bf(f.w) << 16);
  *(uint2*)(out + i) = p;
}

// ---------------- embedding gather ----------------
__global__ void k_embed(const int* __restrict__ x, const float* __restrict__ F,
                        float* __restrict__ emb) {
  int i = (blockIdx.x * 256 + threadIdx.x) * 4;
  int b = i >> 10;
  int k = i & 1023;
  float4 v = *(const float4*)(F + (size_t)x[b] * DIM + k);
  *(float4*)(emb + i) = v;
}

// ---------------- j[b,e] = h[b]·W2_w[e] + W2_b[e]  (wave per output) ----------------
__global__ void k_j(const float* __restrict__ h, const float* __restrict__ W2,
                    const float* __restrict__ W2b, float* __restrict__ jout) {
  int wid  = (blockIdx.x * 256 + threadIdx.x) >> 5;
  int lane = threadIdx.x & 31;
  int b = wid >> 10;
  int e = wid & 1023;
  const float* hv = h + (size_t)b * DIM;
  const float* wv = W2 + (size_t)e * DIM;
  float s = 0.f;
  for (int k = lane; k < DIM; k += 32) s += hv[k] * wv[k];
  s = wave_reduce_add(s);
  if (lane == 0) jout[wid] = s + W2b[e];
}

// ---------------- gates[b,g] = emb·W_ih[g] + h·W_hh[g] + b_ih + b_hh ----------------
__global__ void k_gates(const float* __restrict__ emb, const float* __restrict__ h,
                        const float* __restrict__ Wih, const float* __restrict__ Whh,
                        const float* __restrict__ bih, const float* __restrict__ bhh,
                        float* __restrict__ gates) {
  int wid  = (blockIdx.x * 256 + threadIdx.x) >> 5;
  int lane = threadIdx.x & 31;
  int b = wid >> 12;
  int g = wid & 4095;
  const float* ev = emb + (size_t)b * DIM;
  const float* hv = h   + (size_t)b * DIM;
  const float* wi = Wih + (size_t)g * DIM;
  const float* wh = Whh + (size_t)g * DIM;
  float s = 0.f;
  for (int k = lane; k < DIM; k += 32) s += ev[k] * wi[k] + hv[k] * wh[k];
  s = wave_reduce_add(s);
  if (lane == 0) gates[wid] = s + bih[g] + bhh[g];
}

// ---------------- LSTM elementwise + residual; also emit out in bf16 for fc ----------------
__global__ void k_lstm(const float* __restrict__ gates, const float* __restrict__ c,
                       const float* __restrict__ emb, float* __restrict__ hnew,
                       float* __restrict__ cnew, float* __restrict__ outf,
                       unsigned short* __restrict__ outbf) {
  int i = blockIdx.x * 256 + threadIdx.x;   // b*1024 + e
  int b = i >> 10;
  int e = i & 1023;
  const float* gb = gates + (size_t)b * 4 * DIM;
  float ig = sigmoidf(gb[e]);
  float fg = sigmoidf(gb[DIM + e]);
  float gg = fast_tanh(gb[2 * DIM + e]);
  float og = sigmoidf(gb[3 * DIM + e]);
  float cn = fg * c[i] + ig * gg;
  float hn = og * fast_tanh(cn);
  float o  = hn + emb[i];
  hnew[i] = hn;
  cnew[i] = cn;
  outf[i] = o;
  outbf[i] = f2bf(o);
}

// ---------------- fused attention: tanh(enc_y@W1^T + W1b + j) · V_w  ----------------
#define MT   64     // rows per workgroup
#define LDSK 1032   // padded LDS row stride (ushorts): 516 dwords -> conflict-free

__global__ void __launch_bounds__(256)
k_attn(const float* __restrict__ enc_y, const unsigned short* __restrict__ W1bf,
       const float* __restrict__ W1b, const float* __restrict__ jbuf,
       const float* __restrict__ Vw, const float* __restrict__ Vb,
       float* __restrict__ attL) {
  extern __shared__ char smem[];
  unsigned short* Abuf = (unsigned short*)smem;
  float* rowsum = (float*)(smem + MT * LDSK * sizeof(unsigned short));

  const int tid  = threadIdx.x;
  const int lane = tid & 31;
  const int wave = tid >> 5;
  const int row0 = blockIdx.x * MT;   // flattened [B*S] row base (64 | 512 => no b-crossing)
  const int b    = row0 >> 9;

  if (tid < MT) rowsum[tid] = 0.f;

  // Phase 1: stage enc_y tile (f32 -> bf16) into LDS, coalesced float4 reads
  for (int i = tid; i < MT * DIM / 4; i += 256) {
    int r = (i * 4) >> 10;
    int k = (i * 4) & 1023;
    float4 f = *(const float4*)(enc_y + (size_t)(row0 + r) * DIM + k);
    uint2 p;
    p.x = (unsigned)f2bf(f.x) | ((unsigned)f2bf(f.y) << 16);
    p.y = (unsigned)f2bf(f.z) | ((unsigned)f2bf(f.w) << 16);
    *(uint2*)(Abuf + r * LDSK + k) = p;
  }
  __syncthreads();

  const int half = lane >> 4;     // 0: K-low half, 1: K-high half (contiguous k order, same in A and B)
  const int l15  = lane & 15;

  // Each wave owns N-tiles {wave, wave+8, ...}: W1 streamed once per WG from L2
  for (int nt = wave; nt < DIM / 16; nt += 8) {
    const int   e    = nt * 16 + l15;
    const float w1be = W1b[e];
    const float je   = jbuf[(size_t)b * DIM + e];
    const float vwe  = Vw[e];

    v8f acc[4] = {};
    for (int kt = 0; kt < DIM / 32; ++kt) {
      const int K0 = kt * 32;
      // B frag: B[k][n] = W1[e][k]; lane-half selects K0..K0+15 vs K0+16..K0+31
      v16bf bfr = *(const v16bf*)(W1bf + (size_t)e * DIM + K0 + half * 16);
      #pragma unroll
      for (int m = 0; m < 4; ++m) {
        const unsigned short* ar = Abuf + (m * 16 + l15) * LDSK + K0 + half * 16;
        BF16x16 a;
        a.q[0] = *(const uint4*)(ar);        // K = K0 + 16*half + 0..7
        a.q[1] = *(const uint4*)(ar + 8);    // K = K0 + 16*half + 8..15
        acc[m] = __builtin_amdgcn_wmma_f32_16x16x32_bf16(
            false, a.v, false, bfr, (short)0, acc[m], false, false);
      }
    }

    // Epilogue: bias + j, tanh, ·V_w, reduce 16 e-lanes, cross-wave LDS atomic
    #pragma unroll
    for (int m = 0; m < 4; ++m) {
      #pragma unroll
      for (int rr = 0; rr < 8; ++rr) {
        float v = acc[m][rr] + w1be + je;
        float contrib = vwe * fast_tanh(v);
        contrib += __shfl_xor(contrib, 1, 32);
        contrib += __shfl_xor(contrib, 2, 32);
        contrib += __shfl_xor(contrib, 4, 32);
        contrib += __shfl_xor(contrib, 8, 32);
        if (l15 == 0) atomicAdd(&rowsum[m * 16 + half * 8 + rr], contrib);
      }
    }
  }
  __syncthreads();
  if (tid < MT) attL[row0 + tid] = rowsum[tid] + Vb[0];
}

// ---------------- softmax over S per batch ----------------
__global__ void k_softmax(const float* __restrict__ attL, float* __restrict__ att) {
  __shared__ float red[256];
  int b = blockIdx.x, tid = threadIdx.x;
  const float* L = attL + (size_t)b * SS;
  float x0 = L[tid], x1 = L[tid + 256];
  red[tid] = fmaxf(x0, x1);
  __syncthreads();
  for (int s = 128; s > 0; s >>= 1) {
    if (tid < s) red[tid] = fmaxf(red[tid], red[tid + s]);
    __syncthreads();
  }
  float mx = red[0];
  __syncthreads();
  float e0 = __expf(x0 - mx), e1 = __expf(x1 - mx);
  red[tid] = e0 + e1;
  __syncthreads();
  for (int s = 128; s > 0; s >>= 1) {
    if (tid < s) red[tid] += red[tid + s];
    __syncthreads();
  }
  float inv = 1.f / red[0];
  att[(size_t)b * SS + tid] = e0 * inv;
  att[(size_t)b * SS + tid + 256] = e1 * inv;
}

// ---------------- logits = out @ fc_w^T + fc_b  (wave: 1 N-tile x 4 M-tiles) ----------------
__global__ void __launch_bounds__(256)
k_fc(const unsigned short* __restrict__ outbf, const float* __restrict__ fcw,
     const float* __restrict__ fcb, float* __restrict__ logits) {
  const int tid  = threadIdx.x;
  const int lane = tid & 31;
  const int wave = tid >> 5;
  const int nt   = blockIdx.x * 8 + wave;
  const int half = lane >> 4;
  const int l15  = lane & 15;
  const int n    = nt * 16 + l15;

  v8f acc[4] = {};
  for (int kt = 0; kt < DIM / 32; ++kt) {
    const int K0 = kt * 32;
    // B frag: B[k][n] = fc_w[n][k], convert f32->bf16 in regs (fc_w read exactly once)
    const float* bp = fcw + (size_t)n * DIM + K0 + half * 16;
    BF16x16 bb;
    #pragma unroll
    for (int q = 0; q < 4; ++q) {
      float4 f = *(const float4*)(bp + q * 4);
      bb.u[q * 4 + 0] = f2bf(f.x);
      bb.u[q * 4 + 1] = f2bf(f.y);
      bb.u[q * 4 + 2] = f2bf(f.z);
      bb.u[q * 4 + 3] = f2bf(f.w);
    }
    #pragma unroll
    for (int m = 0; m < 4; ++m) {
      const unsigned short* ar = outbf + (size_t)(m * 16 + l15) * DIM + K0 + half * 16;
      BF16x16 a;
      a.q[0] = *(const uint4*)(ar);
      a.q[1] = *(const uint4*)(ar + 8);
      acc[m] = __builtin_amdgcn_wmma_f32_16x16x32_bf16(
          false, a.v, false, bb.v, (short)0, acc[m], false, false);
    }
  }
  const float bias = fcb[n];
  #pragma unroll
  for (int m = 0; m < 4; ++m) {
    #pragma unroll
    for (int rr = 0; rr < 8; ++rr) {
      int row = m * 16 + half * 8 + rr;
      logits[(size_t)row * VOCAB + n] = acc[m][rr] + bias;
    }
  }
}

extern "C" void kernel_launch(void* const* d_in, const int* in_sizes, int n_in,
                              void* d_out, int out_size, void* d_ws, size_t ws_size,
                              hipStream_t stream) {
  const int*   x    = (const int*)  d_in[0];
  const float* h    = (const float*)d_in[1];
  const float* c    = (const float*)d_in[2];
  const float* ency = (const float*)d_in[3];
  const float* Femb = (const float*)d_in[4];
  const float* W1w  = (const float*)d_in[5];
  const float* W1b  = (const float*)d_in[6];
  const float* W2w  = (const float*)d_in[7];
  const float* W2b  = (const float*)d_in[8];
  const float* Vw   = (const float*)d_in[9];
  const float* Vb   = (const float*)d_in[10];
  const float* Wih  = (const float*)d_in[11];
  const float* Whh  = (const float*)d_in[12];
  const float* bih  = (const float*)d_in[13];
  const float* bhh  = (const float*)d_in[14];
  const float* fcw  = (const float*)d_in[15];
  const float* fcb  = (const float*)d_in[16];

  float* logits = (float*)d_out;                       // [64,32000]
  float* hnew   = logits + (size_t)BB * VOCAB;         // [64,1024]
  float* cnew   = hnew + BB * DIM;                     // [64,1024]
  float* att    = cnew + BB * DIM;                     // [64,512,1]

  char* ws = (char*)d_ws;
  unsigned short* W1bf  = (unsigned short*)(ws + 0);         // 2 MB
  float*          jbuf  = (float*)(ws + 2097152);            // 256 KB
  float*          emb   = (float*)(ws + 2359296);            // 256 KB
  float*          gates = (float*)(ws + 2621440);            // 1 MB
  float*          outf  = (float*)(ws + 3670016);            // 256 KB
  unsigned short* outbf = (unsigned short*)(ws + 3932160);   // 128 KB
  float*          attL  = (float*)(ws + 4063232);            // 128 KB

  k_cvt_w1<<<1024, 256, 0, stream>>>(W1w, W1bf);
  k_embed<<<64, 256, 0, stream>>>(x, Femb, emb);
  k_j<<<8192, 256, 0, stream>>>(h, W2w, W2b, jbuf);
  k_gates<<<32768, 256, 0, stream>>>(emb, h, Wih, Whh, bih, bhh, gates);
  k_lstm<<<256, 256, 0, stream>>>(gates, c, emb, hnew, cnew, outf, outbf);

  const int attn_lds = MT * LDSK * (int)sizeof(unsigned short) + MT * (int)sizeof(float);
  hipFuncSetAttribute((const void*)k_attn, hipFuncAttributeMaxDynamicSharedMemorySize, attn_lds);
  k_attn<<<(BB * SS) / MT, 256, attn_lds, stream>>>(ency, W1bf, W1b, jbuf, Vw, Vb, attL);

  k_softmax<<<BB, 256, 0, stream>>>(attL, att);
  k_fc<<<VOCAB / 16 / 8, 256, 0, stream>>>(outbf, fcw, fcb, logits);
}
